// CrossAttentionFusion_25056839205944
// MI455X (gfx1250) — matrix operand order repaired and checked by reading, not compiled, not run on hardware
//
#include <hip/hip_runtime.h>
#include <stdint.h>

// ---------------------------------------------------------------------------
// CrossAttentionFusion for MI455X (gfx1250): bf16 WMMA, f32 accum.
// Weights are double-buffered in LDS via GLOBAL_LOAD_ASYNC_TO_LDS_B128
// (ASYNCcnt) so WMMA issue never waits on L2 latency; B fragments come from
// LDS (ds_load_b128, bank-conflict-free padded layout).
// ---------------------------------------------------------------------------

#define EDIM 256
#define KSUP 64                      // K-panel size staged per buffer
#define LDW  72                      // padded LDS row pitch in bf16 (144 B)
#define WBUF_BYTES (EDIM * LDW * 2)  // 36864 B per panel buffer

typedef __attribute__((ext_vector_type(16))) __bf16 v16bf;
typedef __attribute__((ext_vector_type(8)))  float  v8f;
typedef __attribute__((ext_vector_type(4)))  float  f32x4;
typedef __attribute__((ext_vector_type(4)))  int    i32x4;

union Frag16 {                 // one WMMA bf16 operand fragment (8 VGPRs)
    v16bf bf;
    i32x4 q[2];
    unsigned u[8];
};

struct ARaw { f32x4 f[4]; };   // 16 raw f32 before bf16 packing

__device__ __forceinline__ unsigned pack2_bf16(float a, float b) {
    unsigned ua = __float_as_uint(a), ub = __float_as_uint(b);
    ua = (ua + 0x7FFFu + ((ua >> 16) & 1u)) >> 16;   // RNE f32->bf16
    ub = (ub + 0x7FFFu + ((ub >> 16) & 1u)) >> 16;
    return ua | (ub << 16);
}
__device__ __forceinline__ unsigned short f2bf(float a) {
    unsigned ua = __float_as_uint(a);
    return (unsigned short)((ua + 0x7FFFu + ((ua >> 16) & 1u)) >> 16);
}

// ---- CDNA5 async global->LDS (ASYNCcnt) ----------------------------------
// Dynamic LDS with no static __shared__ starts at offset 0, so the LDS
// address operand is simply the byte offset into smem.
__device__ __forceinline__ void async_copy16(unsigned lds_byte_off, const void* g) {
    asm volatile("global_load_async_to_lds_b128 %0, %1, off"
                 :: "v"(lds_byte_off),
                    "v"((unsigned long long)(uintptr_t)g)
                 : "memory");
}
__device__ __forceinline__ void wait_async0() {
    asm volatile("s_wait_asynccnt 0x0" ::: "memory");
}

// Stage a 256-row x KSUP-col bf16 weight panel into LDS (row pitch LDW).
// 2048 x 16B chunks spread over the block's threads.
__device__ __forceinline__ void stage_w_async(const unsigned short* W, int ld,
                                              int ksbase, unsigned ldsBase,
                                              int tid, int nthr) {
    for (int c = tid; c < (EDIM * KSUP / 8); c += nthr) {
        const int row = c >> 3;          // 0..255
        const int ch  = c & 7;           // 16B chunk within the 128B row slice
        async_copy16(ldsBase + (unsigned)(row * LDW + ch * 8) * 2,
                     W + (long)row * ld + ksbase + ch * 8);
    }
}

// ---- fragment loads -------------------------------------------------------
__device__ __forceinline__ ARaw loadA_raw_nt(const float* xrow, int kc, int half) {
    const float* p0 = xrow + kc + half * 8;
    ARaw r;
    r.f[0] = __builtin_nontemporal_load((const f32x4*)(p0));
    r.f[1] = __builtin_nontemporal_load((const f32x4*)(p0 + 4));
    r.f[2] = __builtin_nontemporal_load((const f32x4*)(p0 + 16));
    r.f[3] = __builtin_nontemporal_load((const f32x4*)(p0 + 20));
    return r;
}
__device__ __forceinline__ Frag16 packA(const ARaw& r) {
    Frag16 a;
#pragma unroll
    for (int j = 0; j < 4; ++j) {
        a.u[2 * j]     = pack2_bf16(r.f[j].x, r.f[j].y);
        a.u[2 * j + 1] = pack2_bf16(r.f[j].z, r.f[j].w);
    }
    return a;
}
__device__ __forceinline__ Frag16 loadA_bf(const unsigned short* row, int kc, int half) {
    Frag16 a;
    const unsigned short* ap = row + kc + half * 8;
    a.q[0] = *(const i32x4*)(ap);
    a.q[1] = *(const i32x4*)(ap + 16);
    return a;
}
// B fragment from an LDS weight panel: row n, K-offset kk within the panel.
__device__ __forceinline__ Frag16 loadB_lds(const unsigned short* Wl, int kk,
                                            int n, int half) {
    const unsigned short* wp = Wl + n * LDW + kk + half * 8;
    Frag16 b;
    b.q[0] = *(const i32x4*)(wp);      // ds_load_b128
    b.q[1] = *(const i32x4*)(wp + 16);
    return b;
}

// ---------------------------------------------------------------------------
// Kernel 0: f32 -> bf16 weight conversion (tiny; weights then live in L2).
// ---------------------------------------------------------------------------
__global__ void cvt_bf16_kernel(const float* __restrict__ src,
                                unsigned short* __restrict__ dst, int n) {
    int i = (blockIdx.x * blockDim.x + threadIdx.x) * 2;
    if (i < n) {
        *(unsigned*)&dst[i] = pack2_bf16(src[i], src[i + 1]);
    }
}

// ---------------------------------------------------------------------------
// Kernel 1: P = X @ W^T + bias.  256 threads = 8 waves, each wave computes a
// 16-row x 256-col strip (16 WMMA accumulators). Weight panels async-staged
// into double-buffered LDS; A prefetched one k-chunk ahead from global (NT).
// ---------------------------------------------------------------------------
template <int KD>
__global__ __launch_bounds__(256)
void proj_kernel(const float* __restrict__ X,
                 const unsigned short* __restrict__ Wb,
                 const float* __restrict__ bias,
                 float* __restrict__ Pf,
                 unsigned short* __restrict__ Pb) {
    extern __shared__ char smem[];     // 2 weight panel buffers (73728 B)
    unsigned short* wpan = (unsigned short*)smem;

    const int tid  = threadIdx.x;
    const int lane = tid & 31;
    const int wave = tid >> 5;
    const int half = lane >> 4;
    const int l15  = lane & 15;
    const int rowBase = blockIdx.x * 128 + wave * 16;

    v8f acc[16];
    const v8f vzero = {0.f, 0.f, 0.f, 0.f, 0.f, 0.f, 0.f, 0.f};
#pragma unroll
    for (int t = 0; t < 16; ++t) acc[t] = vzero;

    const float* xrow = X + (long)(rowBase + l15) * KD;

    // prologue: stage first weight panel, prefetch first A chunk
    stage_w_async(Wb, KD, 0, 0u, tid, 256);
    ARaw araw = loadA_raw_nt(xrow, 0, half);
    Frag16 afrag = packA(araw);
    wait_async0();
    __syncthreads();

    for (int ks = 0; ks < KD; ks += KSUP) {
        const int  cur  = (ks >> 6) & 1;
        const bool more = (ks + KSUP) < KD;
        if (more) stage_w_async(Wb, KD, ks + KSUP, (unsigned)((cur ^ 1) * WBUF_BYTES), tid, 256);
        const unsigned short* Wl = wpan + cur * (WBUF_BYTES / 2);

#pragma unroll
        for (int kk = 0; kk < KSUP; kk += 32) {
            const int  kc    = ks + kk;
            const bool moreA = (kc + 32) < KD;
            ARaw anext;
            if (moreA) anext = loadA_raw_nt(xrow, kc + 32, half);  // covered by 16 WMMAs
#pragma unroll
            for (int t = 0; t < 16; ++t) {
                Frag16 b = loadB_lds(Wl, kk, t * 16 + l15, half);
                acc[t] = __builtin_amdgcn_wmma_f32_16x16x32_bf16(
                    false, afrag.bf, false, b.bf, (short)0, acc[t], false, false);
            }
            if (moreA) afrag = packA(anext);
        }
        if (more) wait_async0();
        __syncthreads();
    }

    // epilogue: bias, dual-precision store
#pragma unroll
    for (int t = 0; t < 16; ++t) {
        const int n = t * 16 + l15;
        const float bv = bias[n];
#pragma unroll
        for (int i = 0; i < 8; ++i) {
            const int m = rowBase + half * 8 + i;   // C layout: M = 8*half + i
            const float v = acc[t][i] + bv;
            Pf[(long)m * EDIM + n] = v;
            Pb[(long)m * EDIM + n] = f2bf(v);
        }
    }
}

// ---------------------------------------------------------------------------
// Kernel 2: fused attention path + LayerNorm.
//   v   = KV @ Wv^T + bv       (Wv panels async-staged in LDS)
//   att = v  @ Wo^T + bo + Res (v staged bf16 in LDS; Wo panels in LDS)
//   out[:, colOff:+256] = LN(att)
// 128 threads = 4 waves, 64-row tile.
// LDS: 72 KB weight panels + 64 KB v/att staging = 136 KB dynamic.
// ---------------------------------------------------------------------------
__global__ __launch_bounds__(128)
void attn_ln_kernel(const unsigned short* __restrict__ KVb,
                    const unsigned short* __restrict__ Wv,
                    const float* __restrict__ bv,
                    const unsigned short* __restrict__ Wo,
                    const float* __restrict__ bo,
                    const float* __restrict__ Res,
                    const float* __restrict__ ln_g,
                    const float* __restrict__ ln_b,
                    float* __restrict__ Out,
                    int colOff) {
    extern __shared__ char smem[];
    unsigned short* wpan  = (unsigned short*)smem;                  // 2 panels
    unsigned short* vls   = (unsigned short*)(smem + 2 * WBUF_BYTES); // 32 KB
    float*          attls = (float*)(smem + 2 * WBUF_BYTES);          // 64 KB (reuse)

    const int tid  = threadIdx.x;
    const int lane = tid & 31;
    const int wave = tid >> 5;
    const int half = lane >> 4;
    const int l15  = lane & 15;
    const int blkRow  = blockIdx.x * 64;
    const int waveRow = wave * 16;

    const v8f vzero = {0.f, 0.f, 0.f, 0.f, 0.f, 0.f, 0.f, 0.f};
    v8f acc[16];

    // ---------------- GEMM1: v = KV @ Wv^T ----------------
#pragma unroll
    for (int t = 0; t < 16; ++t) acc[t] = vzero;

    const unsigned short* kvrow = KVb + (long)(blkRow + waveRow + l15) * EDIM;

    stage_w_async(Wv, EDIM, 0, 0u, tid, 128);
    wait_async0();
    __syncthreads();

#pragma unroll
    for (int ks = 0; ks < EDIM; ks += KSUP) {
        const int  cur  = (ks >> 6) & 1;
        const bool more = (ks + KSUP) < EDIM;
        if (more) stage_w_async(Wv, EDIM, ks + KSUP, (unsigned)((cur ^ 1) * WBUF_BYTES), tid, 128);
        const unsigned short* Wl = wpan + cur * (WBUF_BYTES / 2);

        Frag16 a0 = loadA_bf(kvrow, ks, half);
        Frag16 a1 = loadA_bf(kvrow, ks + 32, half);
#pragma unroll
        for (int t = 0; t < 16; ++t) {
            Frag16 b = loadB_lds(Wl, 0, t * 16 + l15, half);
            acc[t] = __builtin_amdgcn_wmma_f32_16x16x32_bf16(
                false, a0.bf, false, b.bf, (short)0, acc[t], false, false);
        }
#pragma unroll
        for (int t = 0; t < 16; ++t) {
            Frag16 b = loadB_lds(Wl, 32, t * 16 + l15, half);
            acc[t] = __builtin_amdgcn_wmma_f32_16x16x32_bf16(
                false, a1.bf, false, b.bf, (short)0, acc[t], false, false);
        }
        if (more) wait_async0();
        __syncthreads();
    }

    // stage v (+bias) in LDS as bf16, and pre-stage first Wo panel
#pragma unroll
    for (int t = 0; t < 16; ++t) {
        const int n = t * 16 + l15;
        const float bb = bv[n];
#pragma unroll
        for (int i = 0; i < 8; ++i) {
            const int m = waveRow + half * 8 + i;
            vls[m * EDIM + n] = f2bf(acc[t][i] + bb);
        }
    }
    stage_w_async(Wo, EDIM, 0, 0u, tid, 128);
    wait_async0();
    __syncthreads();

    // ---------------- GEMM2: att = v @ Wo^T ----------------
    v8f acc2[16];
#pragma unroll
    for (int t = 0; t < 16; ++t) acc2[t] = vzero;

    const unsigned short* vrow = vls + (long)(waveRow + l15) * EDIM;
#pragma unroll
    for (int ks = 0; ks < EDIM; ks += KSUP) {
        const int  cur  = (ks >> 6) & 1;
        const bool more = (ks + KSUP) < EDIM;
        if (more) stage_w_async(Wo, EDIM, ks + KSUP, (unsigned)((cur ^ 1) * WBUF_BYTES), tid, 128);
        const unsigned short* Wl = wpan + cur * (WBUF_BYTES / 2);

        Frag16 a0 = loadA_bf(vrow, ks, half);        // ds_load_b128
        Frag16 a1 = loadA_bf(vrow, ks + 32, half);
#pragma unroll
        for (int t = 0; t < 16; ++t) {
            Frag16 b = loadB_lds(Wl, 0, t * 16 + l15, half);
            acc2[t] = __builtin_amdgcn_wmma_f32_16x16x32_bf16(
                false, a0.bf, false, b.bf, (short)0, acc2[t], false, false);
        }
#pragma unroll
        for (int t = 0; t < 16; ++t) {
            Frag16 b = loadB_lds(Wl, 32, t * 16 + l15, half);
            acc2[t] = __builtin_amdgcn_wmma_f32_16x16x32_bf16(
                false, a1.bf, false, b.bf, (short)0, acc2[t], false, false);
        }
        if (more) wait_async0();
        __syncthreads();
    }

    // att + bias + residual (streamed once -> NT) into LDS f32
#pragma unroll
    for (int t = 0; t < 16; ++t) {
        const int n = t * 16 + l15;
        const float bb = bo[n];
#pragma unroll
        for (int i = 0; i < 8; ++i) {
            const int m = waveRow + half * 8 + i;
            const float r = __builtin_nontemporal_load(&Res[(long)(blkRow + m) * EDIM + n]);
            attls[m * EDIM + n] = acc2[t][i] + bb + r;
        }
    }
    __syncthreads();

    // ---------------- LayerNorm, one thread per row ----------------
    if (tid < 64) {
        const int m = tid;
        const float* r = attls + m * EDIM;
        float s = 0.f;
#pragma unroll 8
        for (int n = 0; n < EDIM; ++n) s += r[n];
        const float mean = s * (1.f / EDIM);
        float vs = 0.f;
#pragma unroll 8
        for (int n = 0; n < EDIM; ++n) { const float d = r[n] - mean; vs += d * d; }
        const float inv = rsqrtf(vs * (1.f / EDIM) + 1e-5f);
        float* o = Out + (long)(blkRow + m) * (2 * EDIM) + colOff;
#pragma unroll 8
        for (int n = 0; n < EDIM; ++n)
            __builtin_nontemporal_store((r[n] - mean) * inv * ln_g[n] + ln_b[n], &o[n]);
    }
}

// ---------------------------------------------------------------------------
// Launcher
// ---------------------------------------------------------------------------
extern "C" void kernel_launch(void* const* d_in, const int* in_sizes, int n_in,
                              void* d_out, int out_size, void* d_ws, size_t ws_size,
                              hipStream_t stream) {
    const float* img_embed = (const float*)d_in[0];   // B x 2048
    const float* tab_embed = (const float*)d_in[1];   // B x 128
    const float* Wi        = (const float*)d_in[2];   // 256 x 2048
    const float* bi        = (const float*)d_in[3];
    const float* Wt        = (const float*)d_in[4];   // 256 x 128
    const float* bt        = (const float*)d_in[5];
    const float* in_w_img  = (const float*)d_in[6];   // 768 x 256
    const float* in_b_img  = (const float*)d_in[7];
    const float* out_w_img = (const float*)d_in[8];   // 256 x 256
    const float* out_b_img = (const float*)d_in[9];
    const float* in_w_tab  = (const float*)d_in[10];
    const float* in_b_tab  = (const float*)d_in[11];
    const float* out_w_tab = (const float*)d_in[12];
    const float* out_b_tab = (const float*)d_in[13];
    const float* ln_g      = (const float*)d_in[14];
    const float* ln_b      = (const float*)d_in[15];
    float* out = (float*)d_out;

    const int B = in_sizes[0] / 2048;                 // 16384
    const int E = EDIM;

    // workspace carve-up (256B aligned)
    char* ws = (char*)d_ws;
    size_t off = 0;
    auto take = [&](size_t bytes) -> char* {
        char* p = ws + off;
        off = (off + bytes + 255) & ~(size_t)255;
        return p;
    };
    unsigned short* Wi_b  = (unsigned short*)take((size_t)E * 2048 * 2);
    unsigned short* Wt_b  = (unsigned short*)take((size_t)E * 128 * 2);
    unsigned short* Wvi_b = (unsigned short*)take((size_t)E * E * 2);
    unsigned short* Woi_b = (unsigned short*)take((size_t)E * E * 2);
    unsigned short* Wvt_b = (unsigned short*)take((size_t)E * E * 2);
    unsigned short* Wot_b = (unsigned short*)take((size_t)E * E * 2);
    float*          imgPf = (float*)take((size_t)B * E * 4);
    unsigned short* imgPb = (unsigned short*)take((size_t)B * E * 2);
    float*          tabPf = (float*)take((size_t)B * E * 4);
    unsigned short* tabPb = (unsigned short*)take((size_t)B * E * 2);

    // --- weight conversion f32 -> bf16 ---
    auto cvt = [&](const float* src, unsigned short* dst, int n) {
        int threads = n / 2;
        cvt_bf16_kernel<<<(threads + 255) / 256, 256, 0, stream>>>(src, dst, n);
    };
    cvt(Wi, Wi_b, E * 2048);
    cvt(Wt, Wt_b, E * 128);
    cvt(in_w_img + 2 * E * E, Wvi_b, E * E);   // Wv = in_w[2E:]
    cvt(out_w_img,            Woi_b, E * E);
    cvt(in_w_tab + 2 * E * E, Wvt_b, E * E);
    cvt(out_w_tab,            Wot_b, E * E);

    // --- projections (72 KB LDS: double-buffered weight panels) ---
    const size_t smem_proj = 2 * WBUF_BYTES;
    proj_kernel<2048><<<B / 128, 256, smem_proj, stream>>>(img_embed, Wi_b, bi, imgPf, imgPb);
    proj_kernel<128> <<<B / 128, 256, smem_proj, stream>>>(tab_embed, Wt_b, bt, tabPf, tabPb);

    // --- fused attention + LayerNorm, one launch per path ---
    const size_t smem_attn = 2 * WBUF_BYTES + (size_t)64 * EDIM * sizeof(float);
    attn_ln_kernel<<<B / 64, 128, smem_attn, stream>>>(
        tabPb, Wvi_b, in_b_img + 2 * E, Woi_b, out_b_img, imgPf, ln_g, ln_b, out, 0);
    attn_ln_kernel<<<B / 64, 128, smem_attn, stream>>>(
        imgPb, Wvt_b, in_b_tab + 2 * E, Wot_b, out_b_tab, tabPf, ln_g, ln_b, out, E);
}
